// GRAND_60859686584873
// MI455X (gfx1250) — matrix operand order repaired and checked by reading, not compiled, not run on hardware
//
#include <hip/hip_runtime.h>

#define NFEAT 64
#define NHID  256
#define NCLS  40
#define NCLSP 48          // NCLS padded to multiple of 16
#define TM    128         // rows per workgroup in MLP

typedef __attribute__((ext_vector_type(16))) __bf16 v16bf;
typedef __attribute__((ext_vector_type(8)))  float  v8f;

union ABFrag { unsigned int u[8]; v16bf v; };

__device__ __forceinline__ unsigned short f2bf(float f) {
  unsigned int x = __float_as_uint(f);
  return (unsigned short)((x + 0x7FFFu + ((x >> 16) & 1u)) >> 16);  // RNE
}

// ---------------- graph preprocessing ----------------

__global__ void deg_init_k(float* deg, int n) {
  int i = blockIdx.x * blockDim.x + threadIdx.x;
  if (i < n) deg[i] = 1.0f;                       // self-loop
}

__global__ void deg_edges_k(const int* __restrict__ row, float* deg, int E, int N) {
  int e = blockIdx.x * blockDim.x + threadIdx.x;
  if (e < E) {
    unsigned r = (unsigned)row[e];
    if (r < (unsigned)N) unsafeAtomicAdd(&deg[r], 1.0f);
  }
}

__global__ void dinv_k(float* deg, int n) {      // in-place deg -> deg^{-1/2}
  int i = blockIdx.x * blockDim.x + threadIdx.x;
  if (i < n) deg[i] = rsqrtf(deg[i]);
}

__global__ void ew_k(const int* __restrict__ row, const int* __restrict__ col,
                     const float* __restrict__ dinv, float* ew, int E, int N) {
  int e = blockIdx.x * blockDim.x + threadIdx.x;
  if (e < E) {
    unsigned r = (unsigned)row[e], c = (unsigned)col[e];
    float w = 0.0f;
    if (r < (unsigned)N && c < (unsigned)N) w = dinv[r] * dinv[c];
    ew[e] = w;
  }
}

// h0 = 0.5*x ; y = h0
__global__ void prep_k(const float* __restrict__ x, float* h, float* y, int n) {
  int i = blockIdx.x * blockDim.x + threadIdx.x;
  if (i < n) { float v = 0.5f * x[i]; h[i] = v; y[i] = v; }
}

// hn[i][f] = dinv[i]^2 * h[i][f]   (self-loop contribution; also zero-inits hn)
__global__ void spmm_init_k(const float* __restrict__ h, float* hn,
                            const float* __restrict__ dinv, int n) {
  int i = blockIdx.x * blockDim.x + threadIdx.x;
  if (i < n) {
    float d = dinv[i >> 6];
    hn[i] = d * d * h[i];
  }
}

// hn[row] += ew * h[col]; 32 lanes per edge, float2 per lane (64 feats)
__global__ void __launch_bounds__(256)
spmm_edges_k(const int* __restrict__ row, const int* __restrict__ col,
             const float* __restrict__ ew, const float* __restrict__ h,
             float* hn, int E, int N) {
  int e = blockIdx.x * 8 + (threadIdx.x >> 5);
  if (e >= E) return;
  int lane = threadIdx.x & 31;
  unsigned r = (unsigned)row[e], c = (unsigned)col[e];
  if (r >= (unsigned)N || c >= (unsigned)N) return;
  float w = ew[e];
  float2 hv = *(const float2*)(h + (size_t)c * NFEAT + lane * 2);
  float* dst = hn + (size_t)r * NFEAT + lane * 2;
  unsafeAtomicAdd(dst,     w * hv.x);
  unsafeAtomicAdd(dst + 1, w * hv.y);
}

__global__ void accum_k(float* y, const float* __restrict__ hn, int n) {
  int i = blockIdx.x * blockDim.x + threadIdx.x;
  if (i < n) y[i] += hn[i];
}

// ---------------- fused WMMA MLP ----------------
// A fragment: 16x32 bf16 tile, rows M = lane%16; lanes 0-15 K=0-15,
// lanes 16-31 K=16-31; 2 bf16 per VGPR dword.
__device__ __forceinline__ v16bf load_a(const unsigned short* S, int ld, int m0, int kb) {
  int lane = threadIdx.x & 31;
  int half = lane >> 4;
  const unsigned short* p = S + (size_t)(m0 + (lane & 15)) * ld + kb * 32 + half * 16;
  ABFrag f;
  const unsigned int* q0 = (const unsigned int*)p;
  const unsigned int* q1 = (const unsigned int*)(p + 8);
  f.u[0] = q0[0]; f.u[1] = q0[1]; f.u[2] = q0[2]; f.u[3] = q0[3];
  f.u[4] = q1[0]; f.u[5] = q1[1]; f.u[6] = q1[2]; f.u[7] = q1[3];
  return f.v;
}
// B fragment: 32x16 bf16 tile from N-major (transposed) storage T[N][K];
// cols N = lane%16; lanes 0-15 K=0-15, lanes 16-31 K=16-31.
__device__ __forceinline__ v16bf load_b(const unsigned short* T, int ld, int n0, int kb) {
  int lane = threadIdx.x & 31;
  int khalf = lane >> 4;
  const unsigned short* p = T + (size_t)(n0 + (lane & 15)) * ld + kb * 32 + khalf * 16;
  ABFrag f;
  const unsigned int* q = (const unsigned int*)p;
#pragma unroll
  for (int v = 0; v < 8; ++v) f.u[v] = q[v];
  return f.v;
}

__global__ void __launch_bounds__(256)
mlp_wmma_k(const float* __restrict__ y, const float* __restrict__ W1,
           const float* __restrict__ b1, const float* __restrict__ W2,
           const float* __restrict__ b2, float* __restrict__ out, int N) {
  const int LDY  = NFEAT + 8;   // 72
  const int LDW1 = NFEAT + 8;   // Wt1[NHID][LDW1]
  const int LDH  = NHID + 8;    // 264
  const int LDW2 = NHID + 8;    // Wt2[NCLSP][LDW2]
  extern __shared__ unsigned char smem[];
  unsigned short* sY  = (unsigned short*)smem;           // 128*72
  unsigned short* sW1 = sY + TM * LDY;                   // 256*72 (N-major)
  unsigned short* sH  = sW1 + NHID * LDW1;               // 128*264
  unsigned short* sW2 = sH + TM * LDH;                   // 48*264 (N-major)
  float* sB1 = (float*)(sW2 + NCLSP * LDW2);             // 256
  float* sB2 = sB1 + NHID;                               // 48

  int tid = threadIdx.x;
  long long row0 = (long long)blockIdx.x * TM;
  const float inv9 = 1.0f / 9.0f;

  // stage y tile (scaled by 1/9), W1^T, W2^T (zero-padded), biases
  for (int idx = tid; idx < TM * NFEAT; idx += 256) {
    int m = idx >> 6, k = idx & 63;
    long long r = row0 + m;
    float v = (r < N) ? y[r * NFEAT + k] * inv9 : 0.0f;
    sY[m * LDY + k] = f2bf(v);
  }
  for (int idx = tid; idx < NFEAT * NHID; idx += 256) {
    int k = idx >> 8, n = idx & 255;                     // W1 is [64][256]
    sW1[n * LDW1 + k] = f2bf(W1[idx]);
  }
  for (int idx = tid; idx < NCLSP * NHID; idx += 256) {
    int n = idx / NHID, k = idx % NHID;                  // W2 is [256][40]
    float v = (n < NCLS) ? W2[k * NCLS + n] : 0.0f;
    sW2[n * LDW2 + k] = f2bf(v);
  }
  for (int i = tid; i < NHID; i += 256)  sB1[i] = b1[i];
  for (int i = tid; i < NCLSP; i += 256) sB2[i] = (i < NCLS) ? b2[i] : 0.0f;
  __syncthreads();

  int wave = tid >> 5;
  int lane = tid & 31;

  // layer 1: [128x64] @ [64x256] -> relu -> bf16 into sH
  for (int t = wave; t < (TM / 16) * (NHID / 16); t += 8) {
    int mt = t >> 4, nt = t & 15;
    v8f acc = {};
#pragma unroll
    for (int kb = 0; kb < NFEAT / 32; ++kb) {
      v16bf a = load_a(sY, LDY, mt * 16, kb);
      v16bf b = load_b(sW1, LDW1, nt * 16, kb);
      acc = __builtin_amdgcn_wmma_f32_16x16x32_bf16(false, a, false, b,
                                                    (short)0, acc, false, false);
    }
    int n = nt * 16 + (lane & 15);
    int mbase = mt * 16 + (lane >> 4) * 8;
    float bias = sB1[n];
#pragma unroll
    for (int r = 0; r < 8; ++r) {
      float v = acc[r] + bias;
      v = v > 0.0f ? v : 0.0f;
      sH[(mbase + r) * LDH + n] = f2bf(v);
    }
  }
  __syncthreads();

  // layer 2: [128x256] @ [256x48] -> + b2 -> out (first 40 cols)
  for (int t = wave; t < (TM / 16) * (NCLSP / 16); t += 8) {
    int mt = t / 3, nt = t % 3;
    v8f acc = {};
#pragma unroll
    for (int kb = 0; kb < NHID / 32; ++kb) {
      v16bf a = load_a(sH, LDH, mt * 16, kb);
      v16bf b = load_b(sW2, LDW2, nt * 16, kb);
      acc = __builtin_amdgcn_wmma_f32_16x16x32_bf16(false, a, false, b,
                                                    (short)0, acc, false, false);
    }
    int n = nt * 16 + (lane & 15);
    int mbase = mt * 16 + (lane >> 4) * 8;
    if (n < NCLS) {
      float bias = sB2[n];
#pragma unroll
      for (int r = 0; r < 8; ++r) {
        long long row = row0 + mbase + r;
        if (row < N) out[row * NCLS + n] = acc[r] + bias;
      }
    }
  }
}

// ---------------- launcher ----------------

extern "C" void kernel_launch(void* const* d_in, const int* in_sizes, int n_in,
                              void* d_out, int out_size, void* d_ws, size_t ws_size,
                              hipStream_t stream) {
  const float* x  = (const float*)d_in[0];
  const int*   ei = (const int*)d_in[1];
  const float* W1 = (const float*)d_in[2];
  const float* b1 = (const float*)d_in[3];
  const float* W2 = (const float*)d_in[4];
  const float* b2 = (const float*)d_in[5];
  float* out = (float*)d_out;

  int N = in_sizes[0] / NFEAT;     // 100000
  int E = in_sizes[1] / 2;         // 1200000
  const int* row = ei;
  const int* col = ei + E;

  // workspace carve (256B aligned): dinv[N], ew[E], ha[N*64], hb[N*64], y[N*64]
  auto aln = [](size_t v) { return (v + 255) & ~(size_t)255; };
  unsigned char* w = (unsigned char*)d_ws;
  size_t off = 0;
  float* dinv = (float*)(w + off); off += aln((size_t)N * 4);
  float* ew   = (float*)(w + off); off += aln((size_t)E * 4);
  float* ha   = (float*)(w + off); off += aln((size_t)N * NFEAT * 4);
  float* hb   = (float*)(w + off); off += aln((size_t)N * NFEAT * 4);
  float* yv   = (float*)(w + off);

  int nf = N * NFEAT;
  dim3 b256(256);

  deg_init_k <<<(N + 255) / 256, b256, 0, stream>>>(dinv, N);
  deg_edges_k<<<(E + 255) / 256, b256, 0, stream>>>(row, dinv, E, N);
  dinv_k     <<<(N + 255) / 256, b256, 0, stream>>>(dinv, N);
  ew_k       <<<(E + 255) / 256, b256, 0, stream>>>(row, col, dinv, ew, E, N);
  prep_k     <<<(nf + 255) / 256, b256, 0, stream>>>(x, ha, yv, nf);

  for (int it = 0; it < 8; ++it) {
    spmm_init_k <<<(nf + 255) / 256, b256, 0, stream>>>(ha, hb, dinv, nf);
    spmm_edges_k<<<(E + 7) / 8,     b256, 0, stream>>>(row, col, ew, ha, hb, E, N);
    accum_k     <<<(nf + 255) / 256, b256, 0, stream>>>(yv, hb, nf);
    float* t = ha; ha = hb; hb = t;
  }

  size_t smem = (size_t)(TM * (NFEAT + 8) + NHID * (NFEAT + 8) +
                         TM * (NHID + 8) + NCLSP * (NHID + 8)) * 2 +
                (size_t)(NHID + NCLSP) * 4;   // ~149.4 KB
  mlp_wmma_k<<<(N + TM - 1) / TM, b256, smem, stream>>>(yv, W1, b1, W2, b2, out, N);
}
